// RPEAttention_48481590837612
// MI455X (gfx1250) — matrix operand order repaired and checked by reading, not compiled, not run on hardware
//
#include <hip/hip_runtime.h>
#include <hip/hip_bf16.h>

// ---------------------------------------------------------------------------
// RPE attention for MI455X (gfx1250, wave32, WMMA 16x16x32 bf16).
// Memory-bound (~40 GFLOP vs ~0.5 GB HBM traffic @ 23.3 TB/s): bf16
// activations, all matmuls via v_wmma_f32_16x16x32_bf16, weights/R tensors
// L2-resident, q/k/vT tiles staged to LDS with async-to-LDS when available.
// ---------------------------------------------------------------------------

#define Bq 2
#define Dq 1024
#define Cq 256
#define Tq 32
#define Hq 8
#define Fq 32
#define Eq 1024
#define Mq (Bq * Dq * Tq)   // 65536 GEMM rows

static constexpr float SCALE = 0.17677669529663687f;  // 32^-0.5

typedef __attribute__((ext_vector_type(16))) __bf16 v16bf;
typedef __attribute__((ext_vector_type(8)))  float  v8f;
typedef __attribute__((ext_vector_type(4)))  int    v4i;
typedef unsigned short u16;
typedef __attribute__((ext_vector_type(8)))  u16 u16x8;
typedef __attribute__((ext_vector_type(16))) u16 u16x16;

#define AS1 __attribute__((address_space(1)))
#define AS3 __attribute__((address_space(3)))

// CDNA5 async global->LDS path (guarded: falls back to plain copies)
#ifndef HAVE_ASYNC_LDS
#if defined(__has_builtin)
#if __has_builtin(__builtin_amdgcn_global_load_async_to_lds_b128) && \
    __has_builtin(__builtin_amdgcn_s_wait_asynccnt)
#define HAVE_ASYNC_LDS 1
#endif
#endif
#endif
#ifndef HAVE_ASYNC_LDS
#define HAVE_ASYNC_LDS 0
#endif

__device__ __forceinline__ float bf2f(u16 v) {
    unsigned u = ((unsigned)v) << 16;
    return __builtin_bit_cast(float, u);
}
__device__ __forceinline__ u16 f2bf(float f) {
    unsigned u = __builtin_bit_cast(unsigned, f);
    u += 0x7FFFu + ((u >> 16) & 1u);   // round-to-nearest-even
    return (u16)(u >> 16);
}
__device__ __forceinline__ v16bf cast16(u16x16 v) {
    return __builtin_bit_cast(v16bf, v);
}

// A-fragment (16x32 bf16, row-major source, row owned by lane (l&15),
// half h = l>>4): elems 0..15 -> K = h*16 + j.
__device__ __forceinline__ v16bf load_a_frag(const u16* arow, int h) {
    u16x8 lo = *(const u16x8*)(arow + h * 16);
    u16x8 hi = *(const u16x8*)(arow + h * 16 + 8);
    u16x16 cat = __builtin_shufflevector(lo, hi, 0,1,2,3,4,5,6,7,8,9,10,11,12,13,14,15);
    return cast16(cat);
}

// ---------------------------------------------------------------------------
// K1: GroupNorm(32, C) over [B,D,C,T]; write xn transposed [B,D,T,C]
//     as fp32 (residual) and bf16 (GEMM A operand).
// ---------------------------------------------------------------------------
__global__ __launch_bounds__(256) void groupnorm_kernel(
    const float* __restrict__ x, const float* __restrict__ scale,
    const float* __restrict__ bias, float* __restrict__ xn_f32,
    u16* __restrict__ xn_bf16) {
    const int bd = blockIdx.x;
    const int c  = threadIdx.x;
    const float* xp = x + ((size_t)bd * Cq + c) * Tq;
    float xv[Tq];
    float s = 0.f, sq = 0.f;
#pragma unroll
    for (int t = 0; t < Tq; ++t) { float v = xp[t]; xv[t] = v; s += v; sq += v * v; }
    __shared__ float ssum[Cq], ssq[Cq];
    ssum[c] = s; ssq[c] = sq;
    __syncthreads();
    const int g = c >> 3;  // 8 channels per group
    float gs = 0.f, gq = 0.f;
#pragma unroll
    for (int i = 0; i < 8; ++i) { gs += ssum[g * 8 + i]; gq += ssq[g * 8 + i]; }
    const float mean = gs * (1.0f / 256.0f);
    const float var  = gq * (1.0f / 256.0f) - mean * mean;
    const float inv  = rsqrtf(var + 1e-5f);
    const float sc = scale[c], bi = bias[c];
    float* of = xn_f32 + (size_t)bd * Tq * Cq + c;
    u16*  ob = xn_bf16 + (size_t)bd * Tq * Cq + c;
#pragma unroll
    for (int t = 0; t < Tq; ++t) {
        float v2 = (xv[t] - mean) * inv * sc + bi;
        of[(size_t)t * Cq] = v2;
        ob[(size_t)t * Cq] = f2bf(v2);
    }
}

// ---------------------------------------------------------------------------
// K2: fp32 -> bf16 weight conversion (grid-stride)
// ---------------------------------------------------------------------------
__global__ void f32_to_bf16_kernel(const float* __restrict__ in,
                                   u16* __restrict__ out, int n) {
    for (int i = blockIdx.x * blockDim.x + threadIdx.x; i < n;
         i += gridDim.x * blockDim.x)
        out[i] = f2bf(in[i]);
}

// ---------------------------------------------------------------------------
// K3: QKV GEMM  [M,256] x [768,256]^T -> q (pre-scaled), k, vT  [B,D,H,...]
//     One wave per 16x64 strip: A-fragment reused across 4 N-tiles,
//     K=256 in 8 steps of WMMA 16x16x32.
// ---------------------------------------------------------------------------
__global__ __launch_bounds__(32) void qkv_gemm_kernel(
    const u16* __restrict__ A, const u16* __restrict__ W,
    const float* __restrict__ qkv_b, u16* __restrict__ q,
    u16* __restrict__ k, u16* __restrict__ vt) {
    const int lane = threadIdx.x;
    const int tm = blockIdx.x, tg = blockIdx.y;      // tg covers n = tg*64..+63
    const int ml = lane & 15, h = lane >> 4;
    const u16* arow  = A + (size_t)(tm * 16 + ml) * Cq;
    const u16* wbase = W + (size_t)(tg * 64 + ml) * Cq + h * 16;
    v8f acc[4] = {};
#pragma unroll
    for (int kb = 0; kb < Cq; kb += 32) {
        v16bf a = load_a_frag(arow + kb, h);
#pragma unroll
        for (int j = 0; j < 4; ++j) {
            v16bf b = cast16(*(const u16x16*)(wbase + (size_t)j * 16 * Cq + kb));
            acc[j] = __builtin_amdgcn_wmma_f32_16x16x32_bf16(false, a, false, b,
                                                             (short)0, acc[j],
                                                             false, false);
        }
    }
#pragma unroll
    for (int j = 0; j < 4; ++j) {
#pragma unroll
        for (int r = 0; r < 8; ++r) {
            const int m = tm * 16 + r + 8 * h;
            const int n = tg * 64 + j * 16 + ml;
            float val = acc[j][r] + qkv_b[n];
            const int which = n >> 8, ci = n & 255, hh = ci >> 5, f = ci & 31;
            const int bd = m >> 5, t = m & 31;
            if (which == 0)
                q[(((size_t)bd * Hq + hh) * Tq + t) * Fq + f] = f2bf(val * SCALE);
            else if (which == 1)
                k[(((size_t)bd * Hq + hh) * Tq + t) * Fq + f] = f2bf(val);
            else  // store v transposed: vT[bdh][f][t]
                vt[(((size_t)bd * Hq + hh) * Fq + f) * Tq + t] = f2bf(val);
        }
    }
}

// ---------------------------------------------------------------------------
// K4a: time projection  temb[b,t,:] @ tw[net]^T + tb  -> tp [3,B*T,C]
// ---------------------------------------------------------------------------
__global__ __launch_bounds__(256) void time_proj_kernel(
    const float* __restrict__ temb, const float* __restrict__ tw,
    const float* __restrict__ tb, float* __restrict__ tp) {
    const int blk = blockIdx.x;           // 3*B*T blocks
    const int net = blk / (Bq * Tq);
    const int bt  = blk % (Bq * Tq);
    __shared__ float st[Eq];
    for (int i = threadIdx.x; i < Eq; i += 256) st[i] = temb[(size_t)bt * Eq + i];
    __syncthreads();
    const int co = threadIdx.x;
    const float* w = tw + ((size_t)net * Cq + co) * Eq;
    float acc = tb[net * Cq + co];
    for (int e = 0; e < Eq; e += 4)
        acc += st[e] * w[e] + st[e + 1] * w[e + 1] + st[e + 2] * w[e + 2] +
               st[e + 3] * w[e + 3];
    tp[((size_t)net * (Bq * Tq) + bt) * Cq + co] = acc;
}

// ---------------------------------------------------------------------------
// K4b: R[net,b,t,s,:] = silu(time + dist) @ ow[net]^T + ob   (bf16 output)
// ---------------------------------------------------------------------------
__global__ __launch_bounds__(256) void rpe_r_kernel(
    const float* __restrict__ tp, const float* __restrict__ dw,
    const float* __restrict__ db, const float* __restrict__ ow,
    const float* __restrict__ obias, const int* __restrict__ fi,
    u16* __restrict__ R) {
    const int blk = blockIdx.x;                 // 3*B*T*T blocks
    const int net = blk / (Bq * Tq * Tq);
    const int rem = blk % (Bq * Tq * Tq);
    const int b = rem / (Tq * Tq);
    const int tt = rem % (Tq * Tq);
    const int t = tt >> 5, s = tt & 31;
    const float pd = (float)(fi[b * Tq + t] - fi[b * Tq + s]);
    const float d0 = log1pf(fmaxf(pd, 0.f));
    const float d1 = log1pf(fmaxf(-pd, 0.f));
    const float d2 = (pd == 0.f) ? 1.f : 0.f;
    __shared__ float sl[Cq];
    const int c = threadIdx.x;
    const float* dwr = dw + ((size_t)net * Cq + c) * 3;
    float emb = tp[((size_t)net * (Bq * Tq) + b * Tq + t) * Cq + c] +
                dwr[0] * d0 + dwr[1] * d1 + dwr[2] * d2 + db[net * Cq + c];
    sl[c] = emb / (1.f + __expf(-emb));   // silu
    __syncthreads();
    const float* w = ow + ((size_t)net * Cq + c) * Cq;
    float acc = obias[net * Cq + c];
    for (int k2 = 0; k2 < Cq; k2 += 4)
        acc += sl[k2] * w[k2] + sl[k2 + 1] * w[k2 + 1] + sl[k2 + 2] * w[k2 + 2] +
               sl[k2 + 3] * w[k2 + 3];
    R[(((size_t)net * Bq + b) * Tq * Tq + tt) * Cq + c] = f2bf(acc);
}

// ---------------------------------------------------------------------------
// K5: fused attention per (b,d,h): one wave.
//     q/k/vT staged to LDS (async-to-LDS when available);
//     attn = q.k^T (4 WMMA) + RPE score terms (VALU, L2-resident R),
//     softmax in LDS, out = attn.v (4 WMMA) + R_v term (VALU), bf16 out.
// ---------------------------------------------------------------------------
__global__ __launch_bounds__(32) void attn_kernel(
    const u16* __restrict__ q, const u16* __restrict__ k,
    const u16* __restrict__ vt, const u16* __restrict__ R,
    u16* __restrict__ attn_out) {
    const int lane = threadIdx.x;
    const int bdh = blockIdx.x;
    const int hh = bdh & 7;
    const int bd = bdh >> 3;
    const int b  = bd >> 10;   // D = 1024
    const u16* qb  = q  + (size_t)bdh * Tq * Fq;
    const u16* kb  = k  + (size_t)bdh * Tq * Fq;
    const u16* vtb = vt + (size_t)bdh * Tq * Fq;
    const size_t rnet = (size_t)Bq * Tq * Tq * Cq;
    const u16* Rqp = R + 0 * rnet + (size_t)b * Tq * Tq * Cq + hh * Fq;
    const u16* Rkp = R + 1 * rnet + (size_t)b * Tq * Tq * Cq + hh * Fq;
    const u16* Rvp = R + 2 * rnet + (size_t)b * Tq * Tq * Cq + hh * Fq;
    __builtin_prefetch(Rkp, 0, 1);
    __builtin_prefetch(Rqp, 0, 1);
    __builtin_prefetch(Rvp, 0, 1);

    __shared__ u16 qlds[Tq * Fq];    // 2 KB each
    __shared__ u16 klds[Tq * Fq];
    __shared__ u16 vtlds[Tq * Fq];
    __shared__ float at[Tq][Tq + 1];
    __shared__ float ot[Tq][Fq + 1];

#if HAVE_ASYNC_LDS
    {
        const int o = lane * 64;   // 64 B per lane per tile
        AS1 v4i* gq = (AS1 v4i*)((AS1 char*)(void*)const_cast<u16*>(qb)  + o);
        AS1 v4i* gk = (AS1 v4i*)((AS1 char*)(void*)const_cast<u16*>(kb)  + o);
        AS1 v4i* gv = (AS1 v4i*)((AS1 char*)(void*)const_cast<u16*>(vtb) + o);
        AS3 v4i* lq = (AS3 v4i*)((AS3 char*)(void*)qlds  + o);
        AS3 v4i* lk = (AS3 v4i*)((AS3 char*)(void*)klds  + o);
        AS3 v4i* lv = (AS3 v4i*)((AS3 char*)(void*)vtlds + o);
        __builtin_amdgcn_global_load_async_to_lds_b128(gq, lq, 0, 0);
        __builtin_amdgcn_global_load_async_to_lds_b128(gq, lq, 16, 0);
        __builtin_amdgcn_global_load_async_to_lds_b128(gq, lq, 32, 0);
        __builtin_amdgcn_global_load_async_to_lds_b128(gq, lq, 48, 0);
        __builtin_amdgcn_global_load_async_to_lds_b128(gk, lk, 0, 0);
        __builtin_amdgcn_global_load_async_to_lds_b128(gk, lk, 16, 0);
        __builtin_amdgcn_global_load_async_to_lds_b128(gk, lk, 32, 0);
        __builtin_amdgcn_global_load_async_to_lds_b128(gk, lk, 48, 0);
        __builtin_amdgcn_global_load_async_to_lds_b128(gv, lv, 0, 0);
        __builtin_amdgcn_global_load_async_to_lds_b128(gv, lv, 16, 0);
        __builtin_amdgcn_global_load_async_to_lds_b128(gv, lv, 32, 0);
        __builtin_amdgcn_global_load_async_to_lds_b128(gv, lv, 48, 0);
        __builtin_amdgcn_s_wait_asynccnt(0);
    }
#else
#pragma unroll
    for (int i = lane; i < (Tq * Fq) / 8; i += 32) {
        ((u16x8*)qlds)[i]  = ((const u16x8*)qb)[i];
        ((u16x8*)klds)[i]  = ((const u16x8*)kb)[i];
        ((u16x8*)vtlds)[i] = ((const u16x8*)vtb)[i];
    }
#endif
    __syncthreads();

    // k row `lane` in registers for the R_q term
    float kreg[Fq];
#pragma unroll
    for (int f = 0; f < Fq; ++f) kreg[f] = bf2f(klds[lane * Fq + f]);

    const int ml = lane & 15, h = lane >> 4;

    // attn = q . k^T  (K = F = 32, one WMMA per 16x16 tile)
#pragma unroll
    for (int ti = 0; ti < 2; ++ti) {
        v16bf a = load_a_frag(qlds + (ti * 16 + ml) * Fq, h);
#pragma unroll
        for (int tj = 0; tj < 2; ++tj) {
            // B[k_idx=f][n=s] = k[s][f]; lane owns column s = tj*16 + ml
            v16bf bf = cast16(*(const u16x16*)(klds + (tj * 16 + ml) * Fq + h * 16));
            v8f cc = {};
            cc = __builtin_amdgcn_wmma_f32_16x16x32_bf16(false, a, false, bf, (short)0,
                                                         cc, false, false);
#pragma unroll
            for (int r = 0; r < 8; ++r) at[ti * 16 + r + 8 * h][tj * 16 + ml] = cc[r];
        }
    }
    __syncthreads();

    // RPE score terms: lane owns column s
    {
        const int s = lane;
        for (int t = 0; t < Tq; ++t) {
            const u16* rk = Rkp + ((size_t)t * Tq + s) * Cq;
            const u16* rq = Rqp + ((size_t)s * Tq + t) * Cq;
            float acc = at[t][s];
#pragma unroll
            for (int f = 0; f < Fq; ++f) {
                acc += bf2f(qlds[t * Fq + f]) * bf2f(rk[f]);  // q(scaled) . R_k
                acc += kreg[f] * SCALE * bf2f(rq[f]);         // (k*scale) . R_q ^T
            }
            at[t][s] = acc;
        }
    }
    __syncthreads();

    // softmax over s: lane owns row t
    {
        const int t = lane;
        float mx = -1e30f;
#pragma unroll
        for (int s2 = 0; s2 < Tq; ++s2) mx = fmaxf(mx, at[t][s2]);
        float sum = 0.f;
#pragma unroll
        for (int s2 = 0; s2 < Tq; ++s2) {
            float e = __expf(at[t][s2] - mx);
            at[t][s2] = e;
            sum += e;
        }
        const float inv = 1.0f / sum;
#pragma unroll
        for (int s2 = 0; s2 < Tq; ++s2) at[t][s2] *= inv;
    }
    __syncthreads();

    // out = attn . v  (4 WMMA); A from LDS softmax result, B from vT in LDS
#pragma unroll
    for (int ti = 0; ti < 2; ++ti) {
        u16x16 av;
#pragma unroll
        for (int j = 0; j < 8; ++j) av[j] = f2bf(at[ti * 16 + ml][h * 16 + j]);
#pragma unroll
        for (int j = 0; j < 8; ++j) av[8 + j] = f2bf(at[ti * 16 + ml][h * 16 + 8 + j]);
        v16bf a = cast16(av);
#pragma unroll
        for (int tj = 0; tj < 2; ++tj) {
            // B[k_idx=s][n=f] = v[s][f] = vT[f][s]; lane owns column f
            v16bf bf = cast16(*(const u16x16*)(vtlds + (tj * 16 + ml) * Tq + h * 16));
            v8f cc = {};
            cc = __builtin_amdgcn_wmma_f32_16x16x32_bf16(false, a, false, bf, (short)0,
                                                         cc, false, false);
#pragma unroll
            for (int r = 0; r < 8; ++r) ot[ti * 16 + r + 8 * h][tj * 16 + ml] = cc[r];
        }
    }
    __syncthreads();

    // R_v term + bf16 store; lane owns row t
    {
        const int t = lane;
        float acc[Fq];
#pragma unroll
        for (int f = 0; f < Fq; ++f) acc[f] = ot[t][f];
        for (int s2 = 0; s2 < Tq; ++s2) {
            const float a_ = at[t][s2];
            const u16* rv = Rvp + ((size_t)t * Tq + s2) * Cq;
#pragma unroll
            for (int f = 0; f < Fq; ++f) acc[f] += a_ * bf2f(rv[f]);
        }
        u16* orow = attn_out + ((size_t)bd * Tq + t) * Cq + hh * Fq;
#pragma unroll
        for (int f = 0; f < Fq; ++f) orow[f] = f2bf(acc[f]);
    }
}

// ---------------------------------------------------------------------------
// K6: proj GEMM [M,256]x[256,256]^T + bias + residual; LDS-transposed
//     float4 stores into [B,D,C,T]. One wave per 16x64 strip.
// ---------------------------------------------------------------------------
__global__ __launch_bounds__(32) void proj_gemm_kernel(
    const u16* __restrict__ A, const u16* __restrict__ W,
    const float* __restrict__ proj_b, const float* __restrict__ xn_f32,
    float* __restrict__ out) {
    const int lane = threadIdx.x;
    const int tm = blockIdx.x, tg = blockIdx.y;
    const int ml = lane & 15, h = lane >> 4;
    const u16* arow  = A + (size_t)(tm * 16 + ml) * Cq;
    const u16* wbase = W + (size_t)(tg * 64 + ml) * Cq + h * 16;
    v8f acc[4] = {};
#pragma unroll
    for (int kb = 0; kb < Cq; kb += 32) {
        v16bf a = load_a_frag(arow + kb, h);
#pragma unroll
        for (int j = 0; j < 4; ++j) {
            v16bf b = cast16(*(const u16x16*)(wbase + (size_t)j * 16 * Cq + kb));
            acc[j] = __builtin_amdgcn_wmma_f32_16x16x32_bf16(false, a, false, b,
                                                             (short)0, acc[j],
                                                             false, false);
        }
    }
    // epilogue: bias + residual, transpose 16x64 strip through LDS
    __shared__ float stile[64][17];
#pragma unroll
    for (int j = 0; j < 4; ++j) {
#pragma unroll
        for (int r = 0; r < 8; ++r) {
            const int m = tm * 16 + r + 8 * h;
            const int n = tg * 64 + j * 16 + ml;
            stile[j * 16 + ml][r + 8 * h] =
                acc[j][r] + proj_b[n] + xn_f32[(size_t)m * Cq + n];
        }
    }
    __syncthreads();
    const int bd = tm >> 1, t0 = (tm & 1) * 16, n0 = tg * 64;
#pragma unroll
    for (int i = lane; i < 64 * 4; i += 32) {
        const int row = i >> 2, seg = i & 3;
        float4 vv = { stile[row][seg * 4 + 0], stile[row][seg * 4 + 1],
                      stile[row][seg * 4 + 2], stile[row][seg * 4 + 3] };
        *(float4*)(out + ((size_t)bd * Cq + n0 + row) * Tq + t0 + seg * 4) = vv;
    }
}

// ---------------------------------------------------------------------------
extern "C" void kernel_launch(void* const* d_in, const int* in_sizes, int n_in,
                              void* d_out, int out_size, void* d_ws, size_t ws_size,
                              hipStream_t stream) {
    const float* x          = (const float*)d_in[0];
    const float* temb       = (const float*)d_in[1];
    const float* norm_scale = (const float*)d_in[2];
    const float* norm_bias  = (const float*)d_in[3];
    const float* qkv_w      = (const float*)d_in[4];
    const float* qkv_b      = (const float*)d_in[5];
    const float* proj_w     = (const float*)d_in[6];
    const float* proj_b     = (const float*)d_in[7];
    const float* rpe_dist_w = (const float*)d_in[8];
    const float* rpe_dist_b = (const float*)d_in[9];
    const float* rpe_time_w = (const float*)d_in[10];
    const float* rpe_time_b = (const float*)d_in[11];
    const float* rpe_out_w  = (const float*)d_in[12];
    const float* rpe_out_b  = (const float*)d_in[13];
    const int*   frame_idx  = (const int*)d_in[14];

    // Carve workspace (256B aligned slices)
    size_t off = 0;
    auto alloc = [&](size_t bytes) -> void* {
        off = (off + 255) & ~(size_t)255;
        void* p = (char*)d_ws + off;
        off += bytes;
        return p;
    };
    float* xn_f32  = (float*)alloc((size_t)Mq * Cq * 4);      // 64 MB
    u16*   xn_bf16 = (u16*)alloc((size_t)Mq * Cq * 2);        // 32 MB
    u16*   qbuf    = (u16*)alloc((size_t)Mq * Cq * 2);
    u16*   kbuf    = (u16*)alloc((size_t)Mq * Cq * 2);
    u16*   vtbuf   = (u16*)alloc((size_t)Mq * Cq * 2);
    u16*   attn_o  = (u16*)alloc((size_t)Mq * Cq * 2);
    u16*   wqkv    = (u16*)alloc((size_t)3 * Cq * Cq * 2);
    u16*   wproj   = (u16*)alloc((size_t)Cq * Cq * 2);
    float* tproj   = (float*)alloc((size_t)3 * Bq * Tq * Cq * 4);
    u16*   Rbuf    = (u16*)alloc((size_t)3 * Bq * Tq * Tq * Cq * 2);
    (void)ws_size; (void)n_in; (void)in_sizes; (void)out_size;

    // 1. weight conversion (bf16)
    f32_to_bf16_kernel<<<(3 * Cq * Cq + 255) / 256, 256, 0, stream>>>(qkv_w, wqkv,
                                                                      3 * Cq * Cq);
    f32_to_bf16_kernel<<<(Cq * Cq + 255) / 256, 256, 0, stream>>>(proj_w, wproj,
                                                                  Cq * Cq);
    // 2. GroupNorm -> xn (fp32 + bf16, transposed layout)
    groupnorm_kernel<<<Bq * Dq, 256, 0, stream>>>(x, norm_scale, norm_bias, xn_f32,
                                                  xn_bf16);
    // 3. QKV GEMM (16x64 strip per wave)
    qkv_gemm_kernel<<<dim3(Mq / 16, (3 * Cq) / 64), 32, 0, stream>>>(
        xn_bf16, wqkv, qkv_b, qbuf, kbuf, vtbuf);
    // 4. RPE nets
    time_proj_kernel<<<3 * Bq * Tq, 256, 0, stream>>>(temb, rpe_time_w, rpe_time_b,
                                                      tproj);
    rpe_r_kernel<<<3 * Bq * Tq * Tq, 256, 0, stream>>>(tproj, rpe_dist_w, rpe_dist_b,
                                                       rpe_out_w, rpe_out_b,
                                                       frame_idx, Rbuf);
    // 5. fused attention per (b,d,h)
    attn_kernel<<<Bq * Dq * Hq, 32, 0, stream>>>(qbuf, kbuf, vtbuf, Rbuf, attn_o);
    // 6. proj + bias + residual + transposed float4 store
    proj_gemm_kernel<<<dim3(Mq / 16, Cq / 64), 32, 0, stream>>>(
        attn_o, wproj, proj_b, xn_f32, (float*)d_out);
}
